// ASSRN_44238163149191
// MI455X (gfx1250) — compile-verified, matrix operand order checked by reading, not run on hardware
//
#include <hip/hip_runtime.h>
#include <hip/hip_bf16.h>

typedef _Float16 v16h __attribute__((ext_vector_type(16)));
typedef _Float16 v8h  __attribute__((ext_vector_type(8)));
typedef float    v8f  __attribute__((ext_vector_type(8)));

#define BB   2
#define CCH  64
#define HH   64
#define WWD  64
#define HRR  128
#define QQ   16384
#define HID  256
#define C9   576
#define K1P  608      // 580 padded to multiple of 32
#define KQ   2304
#define EPSF 1e-6f

// Load a 16(M)x32(K) f16 WMMA A/B fragment from row-major memory (row stride in halves).
// Per ISA 7.12.2 (16-bit A-matrix 16x32): lanes 0-15 hold K 0..7 & 16..23 of row M=lane,
// lanes 16-31 hold K 8..15 & 24..31 of row M=lane-16. Two b128 loads per lane.
__device__ __forceinline__ v16h load_frag(const _Float16* base, int strideH, int kb, int lane) {
  const int r  = lane & 15;
  const int hi = (lane >> 4) & 1;
  const _Float16* p = base + (size_t)r * strideH + kb * 32 + hi * 8;
  v8h a = *(const v8h*)(p);
  v8h b = *(const v8h*)(p + 16);
  v16h o;
#pragma unroll
  for (int i = 0; i < 8; ++i) { o[i] = a[i]; o[i + 8] = b[i]; }
  return o;
}

__device__ __forceinline__ v8f wmma32(v16h a, v16h b, v8f c) {
  return __builtin_amdgcn_wmma_f32_16x16x32_f16(false, a, false, b, (short)0, c, false, false);
}

// Pack W [Ksrc x Nsrc] fp32 -> W^T [Npad x Kpad] f16 (zero padded).
__global__ void pack_wT(const float* __restrict__ src, _Float16* __restrict__ dst,
                        int Ksrc, int Nsrc, int Kpad, int Npad) {
  int idx = blockIdx.x * blockDim.x + threadIdx.x;
  int total = Npad * Kpad;
  if (idx >= total) return;
  int n = idx / Kpad, k = idx - n * Kpad;
  float v = (k < Ksrc && n < Nsrc) ? src[(size_t)k * Nsrc + n] : 0.f;
  dst[idx] = (_Float16)v;
}

__global__ __launch_bounds__(128) void assrn_fused(
    const float* __restrict__ x, const float* __restrict__ feature,
    const float* __restrict__ ref, const float* __restrict__ coord,
    const float* __restrict__ cell,
    const float* __restrict__ v_b1, const float* __restrict__ v_b2,
    const float* __restrict__ r_b1, const float* __restrict__ r_b2,
    const float* __restrict__ q_b1, const float* __restrict__ q_b2,
    const _Float16* __restrict__ vW1T, const _Float16* __restrict__ vW2T,
    const _Float16* __restrict__ rW1T, const _Float16* __restrict__ rW2T,
    const _Float16* __restrict__ qW1T, const _Float16* __restrict__ qW2T,
    float* __restrict__ out)
{
  extern __shared__ _Float16 smem[];
  _Float16* sKV  = smem;                 // 16 x 608  key (3x3-unfold gather) + extra
  _Float16* sRF  = sKV + 16 * K1P;       // 16 x 608  bilinear ref sample + extra
  _Float16* sHV  = sRF + 16 * K1P;       // 16 x 256  hidden (v / q reuse)
  _Float16* sHR  = sHV + 16 * HID;       // 16 x 256  hidden (r)
  _Float16* sRES = sHR + 16 * HID;       // 16 x 2304 concatenated preds (q-MLP input)

  const int tid  = threadIdx.x;
  const int lane = tid & 31;
  const int wave = tid >> 5;
  const int blk  = blockIdx.x;
  const int b    = blk >> 10;            // QQ/16 = 1024 tiles per batch
  const int tq   = blk & 1023;

  // cell[:,0,:] scalars -> offset radii
  const float s0 = cell[(size_t)b * QQ * 2 + 0];
  const float s1 = cell[(size_t)b * QQ * 2 + 1];
  const float rx = (1.f - s0) / (HH - 1);
  const float ry = (1.f - s1) / (WWD - 1);

  for (int o = 0; o < 4; ++o) {          // offsets (-1,-1),(-1,1),(1,-1),(1,1)
    __syncthreads();                     // protect sKV/sRF from previous iteration readers
    // ---------------- sampling: fill sKV / sRF ----------------
    {
      const int m   = tid >> 3;          // 16 rows x 8 threads/row
      const int sub = tid & 7;
      const int qg  = tq * 16 + m;
      const size_t qoff = ((size_t)b * QQ + qg) * 2;
      const float c0 = coord[qoff + 0], c1 = coord[qoff + 1];
      const float ce0 = cell[qoff + 0], ce1 = cell[qoff + 1];
      const float vx = (o & 2) ? 1.f : -1.f;
      const float vy = (o & 1) ? 1.f : -1.f;
      float chc = fminf(fmaxf(c0 + vx * rx + EPSF, -1.f + EPSF), 1.f - EPSF);
      float cwc = fminf(fmaxf(c1 + vy * ry + EPSF, -1.f + EPSF), 1.f - EPSF);
      const int ih = (int)fminf(fmaxf(rintf(((chc + 1.f) * HH  - 1.f) * 0.5f), 0.f), (float)(HH - 1));
      const int iw = (int)fminf(fmaxf(rintf(((cwc + 1.f) * WWD - 1.f) * 0.5f), 0.f), (float)(WWD - 1));
      // bilinear setup on HR grid (zeros padding)
      const float fh = ((chc + 1.f) * HRR - 1.f) * 0.5f;
      const float fw = ((cwc + 1.f) * HRR - 1.f) * 0.5f;
      const float h0f = floorf(fh), w0f = floorf(fw);
      const float ah = fh - h0f, aw = fw - w0f;
      const int h0 = (int)h0f, w0 = (int)w0f;
      int   ys[4] = {h0, h0, h0 + 1, h0 + 1};
      int   xs[4] = {w0, w0 + 1, w0, w0 + 1};
      float wt[4] = {(1.f - ah) * (1.f - aw), (1.f - ah) * aw, ah * (1.f - aw), ah * aw};
      int yc[4], xc[4];
#pragma unroll
      for (int j = 0; j < 4; ++j) {
        bool valid = (ys[j] >= 0) && (ys[j] < HRR) && (xs[j] >= 0) && (xs[j] < HRR);
        wt[j] = valid ? wt[j] : 0.f;
        yc[j] = min(max(ys[j], 0), HRR - 1);
        xc[j] = min(max(xs[j], 0), HRR - 1);
      }
      const float ck0 = (2.f * ih + 1.f) / HH  - 1.f;
      const float ck1 = (2.f * iw + 1.f) / WWD - 1.f;
      const float ex[4] = {(c0 - ck0) * HH, (c1 - ck1) * WWD, ce0 * HH, ce1 * WWD};

      const float* fb = feature + (size_t)b * CCH * HH * WWD;
      const float* rb = ref     + (size_t)b * C9 * HRR * HRR;
      for (int c = sub * 76; c < sub * 76 + 76; ++c) {   // 608/8 = 76 cols per thread
        _Float16 kvv, rfv;
        if (c < C9) {
          int cch = c / 9, k = c - cch * 9;
          int y = ih + k / 3 - 1, x2 = iw + (k % 3) - 1;
          float fv = (y >= 0 && y < HH && x2 >= 0 && x2 < WWD)
                       ? fb[((size_t)cch * HH + y) * WWD + x2] : 0.f;
          kvv = (_Float16)fv;
          const float* rc = rb + (size_t)c * HRR * HRR;
          float rv = 0.f;
#pragma unroll
          for (int j = 0; j < 4; ++j) rv += wt[j] * rc[yc[j] * HRR + xc[j]];
          rfv = (_Float16)rv;
        } else if (c < C9 + 4) {
          kvv = rfv = (_Float16)ex[c - C9];
        } else {
          kvv = rfv = (_Float16)0.f;
        }
        sKV[m * K1P + c] = kvv;
        sRF[m * K1P + c] = rfv;
      }
    }
    __syncthreads();

    // ---------------- layer1: hidden = relu(A @ W1^T + b1), v and r ----------------
    for (int nt = wave * 4; nt < wave * 4 + 4; ++nt) {
      v8f accV = {}; v8f accR = {};
      for (int kb = 0; kb < K1P / 32; ++kb) {
        v16h Av = load_frag(sKV, K1P, kb, lane);
        v16h Ar = load_frag(sRF, K1P, kb, lane);
        v16h Bv = load_frag(vW1T + (size_t)nt * 16 * K1P, K1P, kb, lane);
        v16h Br = load_frag(rW1T + (size_t)nt * 16 * K1P, K1P, kb, lane);
        accV = wmma32(Av, Bv, accV);
        accR = wmma32(Ar, Br, accR);
      }
      const int n  = nt * 16 + (lane & 15);
      const int hi = (lane >> 4) & 1;
      const float bv = v_b1[n], br = r_b1[n];
#pragma unroll
      for (int i = 0; i < 8; ++i) {
        int m = i + 8 * hi;                      // D layout: VGPR i -> row i (+8 for hi lanes)
        float hv = accV[i] + bv; hv = hv > 0.f ? hv : 0.f;
        float hr = accR[i] + br; hr = hr > 0.f ? hr : 0.f;
        sHV[m * HID + n] = (_Float16)hv;
        sHR[m * HID + n] = (_Float16)hr;
      }
    }
    __syncthreads();

    // ---------------- layer2 + fused elementwise: res[o] = key*(hV@W2+b) + ref*(hR@W2+b) ----
    v16h hvA[8], hrA[8];
#pragma unroll
    for (int kb = 0; kb < 8; ++kb) {
      hvA[kb] = load_frag(sHV, HID, kb, lane);
      hrA[kb] = load_frag(sHR, HID, kb, lane);
    }
    for (int nt = wave; nt < 36; nt += 4) {      // 9 tiles per wave, uniform trip count
      v8f accV = {}; v8f accR = {};
#pragma unroll
      for (int kb = 0; kb < 8; ++kb) {
        v16h Bv = load_frag(vW2T + (size_t)nt * 16 * HID, HID, kb, lane);
        v16h Br = load_frag(rW2T + (size_t)nt * 16 * HID, HID, kb, lane);
        accV = wmma32(hvA[kb], Bv, accV);
        accR = wmma32(hrA[kb], Br, accR);
      }
      const int n  = nt * 16 + (lane & 15);
      const int hi = (lane >> 4) & 1;
      const float bv = v_b2[n], br = r_b2[n];
#pragma unroll
      for (int i = 0; i < 8; ++i) {
        int m = i + 8 * hi;
        float kvv = (float)sKV[m * K1P + n];
        float rfv = (float)sRF[m * K1P + n];
        float pred = (accV[i] + bv) * kvv + (accR[i] + br) * rfv;  // softmax(singleton)==1
        sRES[m * KQ + o * C9 + n] = (_Float16)pred;
      }
    }
  }
  __syncthreads();

  // ---------------- q-MLP layer1: relu(res @ qW1^T + b1) ----------------
  for (int nt = wave * 4; nt < wave * 4 + 4; ++nt) {
    v8f acc = {};
    for (int kb = 0; kb < KQ / 32; ++kb) {       // 72 K-blocks
      v16h A  = load_frag(sRES, KQ, kb, lane);
      v16h Bf = load_frag(qW1T + (size_t)nt * 16 * KQ, KQ, kb, lane);
      acc = wmma32(A, Bf, acc);
    }
    const int n  = nt * 16 + (lane & 15);
    const int hi = (lane >> 4) & 1;
    const float bq = q_b1[n];
#pragma unroll
    for (int i = 0; i < 8; ++i) {
      int m = i + 8 * hi;
      float hv = acc[i] + bq; hv = hv > 0.f ? hv : 0.f;
      sHV[m * HID + n] = (_Float16)hv;
    }
  }
  __syncthreads();

  // ---------------- q-MLP layer2 (N padded 1->16, col 0 valid) + bilinear(x, border) ----
  if (wave == 0) {
    v8f acc = {};
#pragma unroll
    for (int kb = 0; kb < 8; ++kb) {
      v16h A  = load_frag(sHV, HID, kb, lane);
      v16h Bf = load_frag(qW2T, HID, kb, lane);
      acc = wmma32(A, Bf, acc);
    }
    if ((lane & 15) == 0) {                      // lanes 0 & 16 hold column N=0
      const int hi = (lane >> 4) & 1;
      const float qb = q_b2[0];
      const float* xb = x + (size_t)b * HH * WWD;
#pragma unroll
      for (int i = 0; i < 8; ++i) {
        int m  = i + 8 * hi;
        int qg = tq * 16 + m;
        const size_t qoff = ((size_t)b * QQ + qg) * 2;
        float g0 = coord[qoff + 0], g1 = coord[qoff + 1];
        float fh = fminf(fmaxf(((g0 + 1.f) * HH  - 1.f) * 0.5f, 0.f), (float)(HH - 1));
        float fw = fminf(fmaxf(((g1 + 1.f) * WWD - 1.f) * 0.5f, 0.f), (float)(WWD - 1));
        float h0f = floorf(fh), w0f = floorf(fw);
        float ah = fh - h0f, aw = fw - w0f;
        int h0 = min(max((int)h0f, 0), HH - 1),  w0 = min(max((int)w0f, 0), WWD - 1);
        int h1 = min(h0 + 1, HH - 1),            w1 = min(w0 + 1, WWD - 1);
        float bil = (1.f - ah) * (1.f - aw) * xb[h0 * WWD + w0]
                  + (1.f - ah) * aw         * xb[h0 * WWD + w1]
                  + ah * (1.f - aw)         * xb[h1 * WWD + w0]
                  + ah * aw                 * xb[h1 * WWD + w1];
        out[(size_t)b * QQ + qg] = acc[i] + qb + bil;
      }
    }
  }
}

extern "C" void kernel_launch(void* const* d_in, const int* in_sizes, int n_in,
                              void* d_out, int out_size, void* d_ws, size_t ws_size,
                              hipStream_t stream) {
  const float* x       = (const float*)d_in[0];
  const float* feature = (const float*)d_in[1];
  const float* ref     = (const float*)d_in[2];
  const float* coord   = (const float*)d_in[3];
  const float* cell    = (const float*)d_in[4];
  // d_in[5..8] (k-MLP) are mathematically dead: softmax over a singleton axis == 1.
  const float* v_w1 = (const float*)d_in[9];
  const float* v_b1 = (const float*)d_in[10];
  const float* v_w2 = (const float*)d_in[11];
  const float* v_b2 = (const float*)d_in[12];
  const float* r_w1 = (const float*)d_in[13];
  const float* r_b1 = (const float*)d_in[14];
  const float* r_w2 = (const float*)d_in[15];
  const float* r_b2 = (const float*)d_in[16];
  const float* q_w1 = (const float*)d_in[17];
  const float* q_b1 = (const float*)d_in[18];
  const float* q_w2 = (const float*)d_in[19];
  const float* q_b2 = (const float*)d_in[20];

  _Float16* w    = (_Float16*)d_ws;
  _Float16* vW1T = w;                         // 256 x 608
  _Float16* vW2T = vW1T + 256 * 608;          // 576 x 256
  _Float16* rW1T = vW2T + 576 * 256;          // 256 x 608
  _Float16* rW2T = rW1T + 256 * 608;          // 576 x 256
  _Float16* qW1T = rW2T + 576 * 256;          // 256 x 2304
  _Float16* qW2T = qW1T + 256 * 2304;         // 16  x 256 (rows 1..15 zero)

  pack_wT<<<(256 * 608  + 255) / 256, 256, 0, stream>>>(v_w1, vW1T,  580, 256,  608, 256);
  pack_wT<<<(576 * 256  + 255) / 256, 256, 0, stream>>>(v_w2, vW2T,  256, 576,  256, 576);
  pack_wT<<<(256 * 608  + 255) / 256, 256, 0, stream>>>(r_w1, rW1T,  580, 256,  608, 256);
  pack_wT<<<(576 * 256  + 255) / 256, 256, 0, stream>>>(r_w2, rW2T,  256, 576,  256, 576);
  pack_wT<<<(256 * 2304 + 255) / 256, 256, 0, stream>>>(q_w1, qW1T, 2304, 256, 2304, 256);
  pack_wT<<<(16  * 256  + 255) / 256, 256, 0, stream>>>(q_w2, qW2T,  256,   1,  256,  16);

  const size_t shmem = (size_t)(16 * K1P * 2 + 16 * HID * 2 + 16 * KQ) * sizeof(_Float16); // 129024 B
  assrn_fused<<<BB * (QQ / 16), 128, shmem, stream>>>(
      x, feature, ref, coord, cell,
      v_b1, v_b2, r_b1, r_b2, q_b1, q_b2,
      vW1T, vW2T, rW1T, rW2T, qW1T, qW2T,
      (float*)d_out);
}
